// ResistanceRGCN_56530359550201
// MI455X (gfx1250) — matrix operand order repaired and checked by reading, not compiled, not run on hardware
//
#include <hip/hip_runtime.h>

#define HD __device__ __forceinline__
#define EPS 1e-5f

typedef __attribute__((ext_vector_type(16))) __bf16 v16bf;
typedef __attribute__((ext_vector_type(8)))  float  v8f;

// ---------------------------------------------------------------------------
// WMMA helpers (CDNA5 wave32, v_wmma_f32_16x16x32_bf16)
// ---------------------------------------------------------------------------
HD v8f wmma_bf16(v16bf a, v16bf b, v8f c) {
  return __builtin_amdgcn_wmma_f32_16x16x32_bf16(false, a, false, b, (short)0, c,
                                                 false, false);
}

// A matrix 16x32 bf16 tile, row-major in LDS with row stride lda.
// ISA layout: lanes 0-15 (M=lane) hold K=0..7 (v0-3) and K=16..23 (v4-7);
// lanes 16-31 hold K=8..15 and K=24..31.  -> two contiguous 8-half runs.
HD v16bf load_a_frag(const __bf16* Abase, int lda, int lane) {
  int m = lane & 15, hf = lane >> 4;
  const __bf16* row = Abase + m * lda + hf * 8;
  v16bf r;
#pragma unroll
  for (int i = 0; i < 8; ++i) r[i] = row[i];
#pragma unroll
  for (int i = 0; i < 8; ++i) r[8 + i] = row[16 + i];
  return r;
}

// B matrix 32x16 bf16 tile from K-transposed weights Bt[n][k] (stride ldk).
// ISA layout: lanes 0-15 hold K=0..7 and K=16..23 of column n=lane, lanes 16-31 K=8..15 and K=24..31.
HD v16bf load_bt_frag(const __bf16* Bt, int ldk, int n0, int k0, int lane) {
  const __bf16* col = Bt + (size_t)(n0 + (lane & 15)) * ldk + k0 + (lane >> 4) * 8;
  v16bf r;
#pragma unroll
  for (int i = 0; i < 8; ++i) r[i] = col[i];
#pragma unroll
  for (int i = 0; i < 8; ++i) r[8 + i] = col[16 + i];
  return r;
}

HD void atomicMaxF(float* addr, float v) {
  if (v >= 0.f) atomicMax((int*)addr, __float_as_int(v));
  else          atomicMin((unsigned int*)addr, __float_as_uint(v));
}

// ---------------------------------------------------------------------------
// small utility kernels
// ---------------------------------------------------------------------------
__global__ void k_zero(float* p, int n) {
  int i = blockIdx.x * blockDim.x + threadIdx.x;
  int st = gridDim.x * blockDim.x;
  for (; i < n; i += st) p[i] = 0.f;
}

__global__ void k_f32_to_bf16(const float* s, __bf16* d, int n) {
  int i = blockIdx.x * blockDim.x + threadIdx.x;
  if (i < n) d[i] = (__bf16)s[i];
}

// ---------------------------------------------------------------------------
// Node encoder: h = LN(relu(x @ W_ne + b_ne)); one wave32 per node
// ---------------------------------------------------------------------------
__global__ void __launch_bounds__(32)
k_encode(const float* __restrict__ x, const float* __restrict__ W,
         const float* __restrict__ b, const float* __restrict__ g,
         const float* __restrict__ be, float* __restrict__ h, int N) {
  int n = blockIdx.x;
  if (n >= N) return;
  int lane = threadIdx.x;
  float v[2];
#pragma unroll
  for (int q = 0; q < 2; ++q) {
    int c = lane + q * 32;
    float acc = b[c];
#pragma unroll
    for (int j = 0; j < 16; ++j) acc += x[n * 16 + j] * W[j * 64 + c];
    v[q] = fmaxf(acc, 0.f);
  }
  float s1 = v[0] + v[1], s2 = v[0] * v[0] + v[1] * v[1];
  for (int m = 1; m < 32; m <<= 1) { s1 += __shfl_xor(s1, m, 32); s2 += __shfl_xor(s2, m, 32); }
  float mean = s1 / 64.f, var = s2 / 64.f - mean * mean;
  float rstd = rsqrtf(var + EPS);
#pragma unroll
  for (int q = 0; q < 2; ++q) {
    int c = lane + q * 32;
    h[(size_t)n * 64 + c] = (v[q] - mean) * rstd * g[c] + be[c];
  }
}

// ---------------------------------------------------------------------------
// RGCN scatter: agg[(dst*4+rel)*64+c] += h[src*64+c]; cnt[(dst*4+rel)] += 1
// (mean(msg) = mean(h[src]) @ Wrel since rel is constant per segment)
// ---------------------------------------------------------------------------
__global__ void __launch_bounds__(256)
k_scatter(const int* __restrict__ ei, const float* __restrict__ ea,
          const float* __restrict__ h, float* __restrict__ agg,
          float* __restrict__ cnt, int E) {
  int gid = blockIdx.x * blockDim.x + threadIdx.x;
  if (gid >= E * 64) return;
  int e = gid >> 6, c = gid & 63;
  int src = ei[e], dst = ei[E + e];
  float m0 = ea[e * 8 + 0];
  int rel = 0;
#pragma unroll
  for (int j = 1; j < 4; ++j) {
    float v = ea[e * 8 + j];
    if (v > m0) { m0 = v; rel = j; }
  }
  int seg = dst * 4 + rel;
  atomicAdd(&agg[(size_t)seg * 64 + c], h[(size_t)src * 64 + c]);
  if (c == 0) atomicAdd(&cnt[seg], 1.0f);
}

// ---------------------------------------------------------------------------
// RGCN node update (WMMA): out = [mean_0..mean_3 | h] @ [Wrel;Wroot] + bias,
// then relu(LN(.)) residual-added.  4 waves x 16 nodes per block.
// ---------------------------------------------------------------------------
__global__ void __launch_bounds__(128)
k_rgcn_update(const float* __restrict__ h, const float* __restrict__ agg,
              const float* __restrict__ cnt, const __bf16* __restrict__ Bext,
              const float* __restrict__ bias, const float* __restrict__ g,
              const float* __restrict__ be, float* __restrict__ h_out, int N) {
  __shared__ __bf16 Bt[64 * 320];       // B transposed: Bt[c][k], k=0..319
  __shared__ __bf16 A[4][16 * 320];     // per-wave A tiles
  __shared__ float  outls[64 * 64];
  int tid = threadIdx.x, lane = tid & 31, w = tid >> 5;

  for (int idx = tid; idx < 320 * 64; idx += 128) {
    int k = idx >> 6, c = idx & 63;
    Bt[c * 320 + k] = Bext[idx];
  }
  int nb = blockIdx.x * 64 + w * 16;
  for (int j = lane; j < 16 * 320; j += 32) {
    int m = j / 320, col = j % 320;
    int node = nb + m; if (node >= N) node = N - 1;
    float v;
    if (col < 256) {
      int r = col >> 6, k = col & 63;
      float inv = 1.f / fmaxf(cnt[node * 4 + r], 1.f);
      v = agg[((size_t)(node * 4 + r)) * 64 + k] * inv;
    } else {
      v = h[(size_t)node * 64 + (col - 256)];
    }
    A[w][j] = (__bf16)v;
  }
  __syncthreads();

  v8f acc[4];
#pragma unroll
  for (int i = 0; i < 4; ++i) acc[i] = v8f{0.f, 0.f, 0.f, 0.f, 0.f, 0.f, 0.f, 0.f};
#pragma unroll
  for (int kt = 0; kt < 10; ++kt) {
    v16bf a = load_a_frag(&A[w][kt * 32], 320, lane);
#pragma unroll
    for (int nt = 0; nt < 4; ++nt)
      acc[nt] = wmma_bf16(a, load_bt_frag(Bt, 320, nt * 16, kt * 32, lane), acc[nt]);
  }
  int mh = lane >> 4, nn = lane & 15;
#pragma unroll
  for (int nt = 0; nt < 4; ++nt)
#pragma unroll
    for (int i = 0; i < 8; ++i)
      outls[(w * 16 + mh * 8 + i) * 64 + nt * 16 + nn] = acc[nt][i];
  __syncthreads();

  // LN + relu + residual: 2 lanes per node, 32 channels each
  int m = lane >> 1, part = lane & 1, node = nb + m;
  float vv[32], s1 = 0.f, s2 = 0.f;
#pragma unroll
  for (int j = 0; j < 32; ++j) {
    int c = part * 32 + j;
    float t = outls[(w * 16 + m) * 64 + c] + bias[c];
    vv[j] = t; s1 += t; s2 += t * t;
  }
  s1 += __shfl_xor(s1, 1, 32); s2 += __shfl_xor(s2, 1, 32);
  float mean = s1 / 64.f, var = s2 / 64.f - mean * mean, rstd = rsqrtf(var + EPS);
  if (node < N) {
#pragma unroll
    for (int j = 0; j < 32; ++j) {
      int c = part * 32 + j;
      float t = fmaxf((vv[j] - mean) * rstd * g[c] + be[c], 0.f);
      h_out[(size_t)node * 64 + c] = h[(size_t)node * 64 + c] + t;
    }
  }
}

// ---------------------------------------------------------------------------
// Attention pooling kernels
// ---------------------------------------------------------------------------
__global__ void k_pool_init(float* gmax, float* denom, float* gemb) {
  int t = blockIdx.x * blockDim.x + threadIdx.x;
  if (t < 64) { gmax[t] = -3.402823466e38f; denom[t] = 0.f; }
  if (t < 64 * 64) gemb[t] = 0.f;
}

__global__ void __launch_bounds__(256)
k_gate(const float* __restrict__ h, const int* __restrict__ batch,
       const float* __restrict__ wg, const float* __restrict__ bg,
       float* __restrict__ gate, float* __restrict__ gmax, int N) {
  int n = blockIdx.x * blockDim.x + threadIdx.x;
  if (n >= N) return;
  float acc = bg[0];
#pragma unroll
  for (int c = 0; c < 64; ++c) acc += h[(size_t)n * 64 + c] * wg[c];
  gate[n] = acc;
  atomicMaxF(&gmax[batch[n]], acc);
}

__global__ void __launch_bounds__(256)
k_exp(float* __restrict__ gate, const int* __restrict__ batch,
      const float* __restrict__ gmax, float* __restrict__ denom, int N) {
  int n = blockIdx.x * blockDim.x + threadIdx.x;
  if (n >= N) return;
  float e = __expf(gate[n] - gmax[batch[n]]);
  gate[n] = e;
  atomicAdd(&denom[batch[n]], e);
}

__global__ void __launch_bounds__(256)
k_accum(const float* __restrict__ h, const int* __restrict__ batch,
        const float* __restrict__ e, float* __restrict__ gemb, int N) {
  int gid = blockIdx.x * blockDim.x + threadIdx.x;
  if (gid >= N * 64) return;
  int n = gid >> 6, c = gid & 63;
  atomicAdd(&gemb[batch[n] * 64 + c], h[(size_t)n * 64 + c] * e[n]);
}

__global__ void k_gfin(float* gemb, const float* denom) {
  int t = blockIdx.x * blockDim.x + threadIdx.x;
  if (t < 64 * 64) gemb[t] /= denom[t >> 6];
}

// ---------------------------------------------------------------------------
// Fused edge MLP: ef=[h[src]|h[dst]|edge_emb|gemb[batch[src]]] (bf16 LDS tile),
// att = sigmoid(relu(ef@Wa+b)@Wa2+b), trans = LN(relu(ef@Wt+b)),
// hmid = LN(relu((trans*att)@Wm+b)), out = hmid@Wm2+b.  4 waves x 16 edges.
// ---------------------------------------------------------------------------
__global__ void __launch_bounds__(128)
k_edge_mlp(const int* __restrict__ ei, const float* __restrict__ ea,
           const int* __restrict__ batch, const float* __restrict__ h,
           const float* __restrict__ gemb,
           const __bf16* __restrict__ Wa, const __bf16* __restrict__ Wt,
           const __bf16* __restrict__ Wm,
           const float* __restrict__ W_ee, const float* __restrict__ b_ee,
           const float* __restrict__ g_ee, const float* __restrict__ be_ee,
           const float* __restrict__ b_a1, const float* __restrict__ W_a2,
           const float* __restrict__ b_a2, const float* __restrict__ b_t1,
           const float* __restrict__ g_t, const float* __restrict__ be_t,
           const float* __restrict__ b_m1, const float* __restrict__ g_m,
           const float* __restrict__ be_m, const float* __restrict__ W_m2,
           const float* __restrict__ b_m2, float* __restrict__ out, int E) {
  __shared__ __bf16 WaT[64 * 256];   // Wa transposed: [n][k]
  __shared__ __bf16 WtT[64 * 256];
  __shared__ __bf16 WmT[32 * 64];
  __shared__ __bf16 ef[4][16 * 256];
  __shared__ __bf16 abuf[4][16 * 64];
  int tid = threadIdx.x, lane = tid & 31, w = tid >> 5;

  for (int idx = tid; idx < 256 * 64; idx += 128) {
    int k = idx >> 6, c = idx & 63;
    WaT[c * 256 + k] = Wa[idx];
    WtT[c * 256 + k] = Wt[idx];
  }
  for (int idx = tid; idx < 64 * 32; idx += 128) {
    int k = idx >> 5, c = idx & 31;
    WmT[c * 64 + k] = Wm[idx];
  }

  int eb = blockIdx.x * 64 + w * 16;
  {  // stage ef tile: 2 lanes per edge, 32 channels each
    int m = lane >> 1, part = lane & 1;
    int e = eb + m; if (e >= E) e = E - 1;
    int src = ei[e], dst = ei[E + e], gi = batch[src];
#pragma unroll
    for (int j = 0; j < 32; ++j) {
      int c = part * 32 + j;
      ef[w][m * 256 + c]       = (__bf16)h[(size_t)src * 64 + c];
      ef[w][m * 256 + 64 + c]  = (__bf16)h[(size_t)dst * 64 + c];
      ef[w][m * 256 + 192 + c] = (__bf16)gemb[gi * 64 + c];
    }
    float ein[8];
#pragma unroll
    for (int j = 0; j < 8; ++j) ein[j] = ea[(size_t)e * 8 + j];
    float vv[32], s1 = 0.f, s2 = 0.f;
#pragma unroll
    for (int j = 0; j < 32; ++j) {
      int c = part * 32 + j;
      float acc = b_ee[c];
#pragma unroll
      for (int q = 0; q < 8; ++q) acc += ein[q] * W_ee[q * 64 + c];
      acc = fmaxf(acc, 0.f);
      vv[j] = acc; s1 += acc; s2 += acc * acc;
    }
    s1 += __shfl_xor(s1, 1, 32); s2 += __shfl_xor(s2, 1, 32);
    float mean = s1 / 64.f, var = s2 / 64.f - mean * mean, rstd = rsqrtf(var + EPS);
#pragma unroll
    for (int j = 0; j < 32; ++j) {
      int c = part * 32 + j;
      ef[w][m * 256 + 128 + c] = (__bf16)((vv[j] - mean) * rstd * g_ee[c] + be_ee[c]);
    }
  }
  __syncthreads();

  int mh = lane >> 4, nn = lane & 15;

  // ---- GEMM A: y = relu(ef@Wa + b_a1); att = sigmoid(y@W_a2 + b_a2) ----
  v8f accA[4];
#pragma unroll
  for (int i = 0; i < 4; ++i) accA[i] = v8f{0.f, 0.f, 0.f, 0.f, 0.f, 0.f, 0.f, 0.f};
#pragma unroll
  for (int kt = 0; kt < 8; ++kt) {
    v16bf a = load_a_frag(&ef[w][kt * 32], 256, lane);
#pragma unroll
    for (int nt = 0; nt < 4; ++nt)
      accA[nt] = wmma_bf16(a, load_bt_frag(WaT, 256, nt * 16, kt * 32, lane), accA[nt]);
  }
  float attp[8] = {0.f, 0.f, 0.f, 0.f, 0.f, 0.f, 0.f, 0.f};
#pragma unroll
  for (int nt = 0; nt < 4; ++nt) {
    int c = nt * 16 + nn;
#pragma unroll
    for (int i = 0; i < 8; ++i) {
      float y = fmaxf(accA[nt][i] + b_a1[c], 0.f);
      attp[i] += y * W_a2[c];
    }
  }
#pragma unroll
  for (int i = 0; i < 8; ++i)
    for (int msk = 1; msk < 16; msk <<= 1) attp[i] += __shfl_xor(attp[i], msk, 32);
  float att[8];
#pragma unroll
  for (int i = 0; i < 8; ++i) att[i] = 1.f / (1.f + __expf(-(attp[i] + b_a2[0])));

  // ---- GEMM T: trans = LN(relu(ef@Wt + b_t1)); attended = trans*att ----
  v8f accT[4];
#pragma unroll
  for (int i = 0; i < 4; ++i) accT[i] = v8f{0.f, 0.f, 0.f, 0.f, 0.f, 0.f, 0.f, 0.f};
#pragma unroll
  for (int kt = 0; kt < 8; ++kt) {
    v16bf a = load_a_frag(&ef[w][kt * 32], 256, lane);
#pragma unroll
    for (int nt = 0; nt < 4; ++nt)
      accT[nt] = wmma_bf16(a, load_bt_frag(WtT, 256, nt * 16, kt * 32, lane), accT[nt]);
  }
  float z[4][8], s1[8], s2[8];
#pragma unroll
  for (int i = 0; i < 8; ++i) { s1[i] = 0.f; s2[i] = 0.f; }
#pragma unroll
  for (int nt = 0; nt < 4; ++nt) {
    int c = nt * 16 + nn;
#pragma unroll
    for (int i = 0; i < 8; ++i) {
      float t = fmaxf(accT[nt][i] + b_t1[c], 0.f);
      z[nt][i] = t; s1[i] += t; s2[i] += t * t;
    }
  }
#pragma unroll
  for (int i = 0; i < 8; ++i)
    for (int msk = 1; msk < 16; msk <<= 1) {
      s1[i] += __shfl_xor(s1[i], msk, 32);
      s2[i] += __shfl_xor(s2[i], msk, 32);
    }
  float mean[8], rstd[8];
#pragma unroll
  for (int i = 0; i < 8; ++i) {
    mean[i] = s1[i] / 64.f;
    float var = s2[i] / 64.f - mean[i] * mean[i];
    rstd[i] = rsqrtf(var + EPS);
  }
#pragma unroll
  for (int nt = 0; nt < 4; ++nt) {
    int c = nt * 16 + nn;
#pragma unroll
    for (int i = 0; i < 8; ++i) {
      float tn = (z[nt][i] - mean[i]) * rstd[i] * g_t[c] + be_t[c];
      abuf[w][(mh * 8 + i) * 64 + c] = (__bf16)(tn * att[i]);
    }
  }
  asm volatile("s_wait_dscnt 0" ::: "memory");

  // ---- GEMM M: hmid = LN(relu(attended@Wm + b_m1)); out = hmid@W_m2+b ----
  v8f accM[2];
#pragma unroll
  for (int i = 0; i < 2; ++i) accM[i] = v8f{0.f, 0.f, 0.f, 0.f, 0.f, 0.f, 0.f, 0.f};
#pragma unroll
  for (int kt = 0; kt < 2; ++kt) {
    v16bf a = load_a_frag(&abuf[w][kt * 32], 64, lane);
#pragma unroll
    for (int nt = 0; nt < 2; ++nt)
      accM[nt] = wmma_bf16(a, load_bt_frag(WmT, 64, nt * 16, kt * 32, lane), accM[nt]);
  }
  float u[2][8], t1[8], t2[8];
#pragma unroll
  for (int i = 0; i < 8; ++i) { t1[i] = 0.f; t2[i] = 0.f; }
#pragma unroll
  for (int nt = 0; nt < 2; ++nt) {
    int c = nt * 16 + nn;
#pragma unroll
    for (int i = 0; i < 8; ++i) {
      float t = fmaxf(accM[nt][i] + b_m1[c], 0.f);
      u[nt][i] = t; t1[i] += t; t2[i] += t * t;
    }
  }
#pragma unroll
  for (int i = 0; i < 8; ++i)
    for (int msk = 1; msk < 16; msk <<= 1) {
      t1[i] += __shfl_xor(t1[i], msk, 32);
      t2[i] += __shfl_xor(t2[i], msk, 32);
    }
  float op[8] = {0.f, 0.f, 0.f, 0.f, 0.f, 0.f, 0.f, 0.f};
#pragma unroll
  for (int i = 0; i < 8; ++i) {
    float mn = t1[i] / 32.f;
    float rs = rsqrtf(t2[i] / 32.f - mn * mn + EPS);
#pragma unroll
    for (int nt = 0; nt < 2; ++nt) {
      int c = nt * 16 + nn;
      float hm = (u[nt][i] - mn) * rs * g_m[c] + be_m[c];
      op[i] += hm * W_m2[c];
    }
  }
#pragma unroll
  for (int i = 0; i < 8; ++i)
    for (int msk = 1; msk < 16; msk <<= 1) op[i] += __shfl_xor(op[i], msk, 32);

  if (nn == 0) {
#pragma unroll
    for (int i = 0; i < 8; ++i) {
      int e = eb + mh * 8 + i;
      if (e < E) out[e] = op[i] + b_m2[0];
    }
  }
}

// ---------------------------------------------------------------------------
// host launcher
// ---------------------------------------------------------------------------
extern "C" void kernel_launch(void* const* d_in, const int* in_sizes, int n_in,
                              void* d_out, int out_size, void* d_ws, size_t ws_size,
                              hipStream_t stream) {
  const float* x     = (const float*)d_in[0];
  const int*   ei    = (const int*)d_in[1];
  const float* ea    = (const float*)d_in[2];
  const int*   batch = (const int*)d_in[3];
  const float* W_ne  = (const float*)d_in[4];
  const float* b_ne  = (const float*)d_in[5];
  const float* g_ne  = (const float*)d_in[6];
  const float* be_ne = (const float*)d_in[7];
  const float *Wrel[3], *Wroot[3], *bL[3], *gL[3], *beL[3];
  for (int L = 0; L < 3; ++L) {
    int base = 8 + L * 5;
    Wrel[L]  = (const float*)d_in[base + 0];
    Wroot[L] = (const float*)d_in[base + 1];
    bL[L]    = (const float*)d_in[base + 2];
    gL[L]    = (const float*)d_in[base + 3];
    beL[L]   = (const float*)d_in[base + 4];
  }
  const float* W_ee  = (const float*)d_in[23];
  const float* b_ee  = (const float*)d_in[24];
  const float* g_ee  = (const float*)d_in[25];
  const float* be_ee = (const float*)d_in[26];
  const float* w_gate= (const float*)d_in[27];
  const float* b_gate= (const float*)d_in[28];
  const float* W_a1  = (const float*)d_in[29];
  const float* b_a1  = (const float*)d_in[30];
  const float* W_a2  = (const float*)d_in[31];
  const float* b_a2  = (const float*)d_in[32];
  const float* W_t1  = (const float*)d_in[33];
  const float* b_t1  = (const float*)d_in[34];
  const float* g_t   = (const float*)d_in[35];
  const float* be_t  = (const float*)d_in[36];
  const float* W_m1  = (const float*)d_in[37];
  const float* b_m1  = (const float*)d_in[38];
  const float* g_m   = (const float*)d_in[39];
  const float* be_m  = (const float*)d_in[40];
  const float* W_m2  = (const float*)d_in[41];
  const float* b_m2  = (const float*)d_in[42];

  const int N = in_sizes[3];       // batch has N entries
  const int E = in_sizes[2] / 8;   // edge_attr is [E,8]

  // workspace carving
  char* ws = (char*)d_ws;
  size_t off = 0;
  auto carve = [&](size_t bytes) -> char* {
    char* p = ws + off;
    off += (bytes + 255) & ~(size_t)255;
    return p;
  };
  float* h_a  = (float*)carve((size_t)N * 64 * 4);
  float* h_b  = (float*)carve((size_t)N * 64 * 4);
  float* agg  = (float*)carve((size_t)N * 4 * 64 * 4);
  float* cnt  = (float*)carve((size_t)N * 4 * 4);
  float* gate = (float*)carve((size_t)N * 4);
  float* gmax = (float*)carve(64 * 4);
  float* denom= (float*)carve(64 * 4);
  float* gemb = (float*)carve(64 * 64 * 4);
  __bf16* Bext[3];
  for (int L = 0; L < 3; ++L) Bext[L] = (__bf16*)carve(320 * 64 * 2);
  __bf16* WaB = (__bf16*)carve(256 * 64 * 2);
  __bf16* WtB = (__bf16*)carve(256 * 64 * 2);
  __bf16* WmB = (__bf16*)carve(64 * 32 * 2);

  auto cv = [&](const float* s, __bf16* d, int n) {
    k_f32_to_bf16<<<(n + 255) / 256, 256, 0, stream>>>(s, d, n);
  };
  // B_ext = [Wrel(4x64x64 rows 0..255); Wroot(rows 256..319)], already row-major [K][64]
  for (int L = 0; L < 3; ++L) {
    cv(Wrel[L], Bext[L], 4 * 64 * 64);
    cv(Wroot[L], Bext[L] + 4 * 64 * 64, 64 * 64);
  }
  cv(W_a1, WaB, 256 * 64);
  cv(W_t1, WtB, 256 * 64);
  cv(W_m1, WmB, 64 * 32);

  k_encode<<<N, 32, 0, stream>>>(x, W_ne, b_ne, g_ne, be_ne, h_a, N);

  float* hin = h_a;
  float* hout = h_b;
  for (int L = 0; L < 3; ++L) {
    k_zero<<<2048, 256, 0, stream>>>(agg, N * 4 * 64);
    k_zero<<<256, 256, 0, stream>>>(cnt, N * 4);
    int tot = E * 64;
    k_scatter<<<(tot + 255) / 256, 256, 0, stream>>>(ei, ea, hin, agg, cnt, E);
    k_rgcn_update<<<(N + 63) / 64, 128, 0, stream>>>(hin, agg, cnt, Bext[L], bL[L],
                                                     gL[L], beL[L], hout, N);
    float* t = hin; hin = hout; hout = t;
  }
  const float* hf = hin;  // final node features (ends in h_b after 3 swaps)

  k_pool_init<<<16, 256, 0, stream>>>(gmax, denom, gemb);
  k_gate<<<(N + 255) / 256, 256, 0, stream>>>(hf, batch, w_gate, b_gate, gate, gmax, N);
  k_exp<<<(N + 255) / 256, 256, 0, stream>>>(gate, batch, gmax, denom, N);
  k_accum<<<(N * 64 + 255) / 256, 256, 0, stream>>>(hf, batch, gate, gemb, N);
  k_gfin<<<16, 256, 0, stream>>>(gemb, denom);

  k_edge_mlp<<<(E + 63) / 64, 128, 0, stream>>>(
      ei, ea, batch, hf, gemb, WaB, WtB, WmB,
      W_ee, b_ee, g_ee, be_ee,
      b_a1, W_a2, b_a2, b_t1, g_t, be_t,
      b_m1, g_m, be_m, W_m2, b_m2,
      (float*)d_out, E);
}
